// STL_3307124818331
// MI455X (gfx1250) — matrix operand order repaired and checked by reading, not compiled, not run on hardware
//
#include <hip/hip_runtime.h>
#include <cstdint>
#include <cstddef>

// ---------------------------------------------------------------------------
// Swin block for MI455X (gfx1250): all GEMMs on v_wmma_f32_16x16x32_bf16,
// fp32 accumulation, fp32 softmax + residuals.
// Data movement: TDM (tensor_load_to_lds, 6-arg amdgpu-toolchain form) for
// GEMM A-tiles, async global->LDS for attention q/k tiles.
// ---------------------------------------------------------------------------

typedef __attribute__((ext_vector_type(16))) __bf16 v16bf;
typedef __attribute__((ext_vector_type(8)))  float  v8f;
typedef __attribute__((ext_vector_type(4)))  unsigned v4u;
typedef __attribute__((ext_vector_type(4)))  int   v4i;
typedef __attribute__((ext_vector_type(8)))  int   v8i;

#if defined(__HIP_DEVICE_COMPILE__) && __has_builtin(__builtin_amdgcn_tensor_load_to_lds) && __has_builtin(__builtin_amdgcn_s_wait_tensorcnt)
#define USE_TDM 1
#else
#define USE_TDM 0
#endif

#if defined(__HIP_DEVICE_COMPILE__) && __has_builtin(__builtin_amdgcn_global_load_async_to_lds_b128) && __has_builtin(__builtin_amdgcn_s_wait_asynccnt)
#define USE_ASYNC 1
#else
#define USE_ASYNC 0
#endif

#if defined(__HIP_DEVICE_COMPILE__)
typedef __attribute__((address_space(3))) void lds_void;

__device__ __forceinline__ unsigned lds_addr32(void* p) {
  // generic -> LDS addrspace cast, then ptrtoint: byte offset in workgroup LDS
  return (unsigned)(unsigned long long)(lds_void*)p;
}
#endif

union Frag16 { v16bf v; uint4 u[2]; };

__device__ __forceinline__ __bf16 f2bf(float f) {
  unsigned u; __builtin_memcpy(&u, &f, 4);
  unsigned r = u + 0x7FFFu + ((u >> 16) & 1u);   // round-to-nearest-even
  unsigned short h = (unsigned short)(r >> 16);
  __bf16 b; __builtin_memcpy(&b, &h, 2);
  return b;
}

#if USE_TDM
// Issue one TDM 2D tile load: rows x cols bf16, row stride (elements) in
// global memory; LDS rows padded by pad_amount DWORDs every pad_interval.
__device__ __forceinline__ void tdm_load_2d_bf16(const __bf16* gsrc,
                                                 unsigned lds_byte_off,
                                                 int rows, int cols,
                                                 int row_stride_elems)
{
  unsigned long long ga = (unsigned long long)gsrc;
  v4u g0;
  g0[0] = 1u;                                   // count=1, user descriptor
  g0[1] = lds_byte_off;                         // lds_addr
  g0[2] = (unsigned)(ga & 0xFFFFFFFFull);       // global_addr[31:0]
  g0[3] = (unsigned)((ga >> 32) & 0x01FFFFFFull) | (2u << 30); // [56:32] | type=2
  v8i g1;
  // data_size=1 (2B), pad_enable, pad_interval code 3 (=16 DWORDs = 64B row),
  // pad_amount code 3 (=4 DWORDs = 16B) -> LDS row pitch 80B (LDA=40 elems)
  g1[0] = (int)((1u << 16) | (1u << 20) | (3u << 22) | (3u << 25));
  g1[1] = (int)(((unsigned)cols & 0xFFFFu) << 16);            // tensor_dim0 lo
  g1[2] = (int)((((unsigned)cols >> 16) & 0xFFFFu) |
                (((unsigned)rows & 0xFFFFu) << 16));          // dim0 hi | dim1 lo
  g1[3] = (int)((((unsigned)rows >> 16) & 0xFFFFu) |
                (((unsigned)cols & 0xFFFFu) << 16));          // dim1 hi | tile_dim0
  g1[4] = (int)((unsigned)rows & 0xFFFFu);                    // tile_dim1 (tile_dim2=0)
  g1[5] = row_stride_elems;                                   // tensor_dim0_stride lo32
  g1[6] = 0;
  g1[7] = 0;
  v4i z4 = {0, 0, 0, 0};
  v8i z8 = {0, 0, 0, 0, 0, 0, 0, 0};
  __builtin_amdgcn_tensor_load_to_lds(g0, g1, z4, z4, z8, 0);
}
#endif

// ---------------------------------------------------------------------------
// Generic bf16 GEMM: C = A(MxK) * B(KxN) + bias, with per-use epilogues.
// Block = 256 threads (8 wave32), tile 128x64, K-step 32.
// Each wave: 2x2 grid of 16x16 WMMA tiles (4 v_wmma per K-step).
// ---------------------------------------------------------------------------
constexpr int BM = 128, BN = 64, BK = 32;
constexpr int LDA = 40;  // bf16 elems per As row (80B; TDM pad reproduces this)
constexpr int LDB = 40;  // bf16 elems per Bs row (B stored transposed: [n][k])

// EPI: 0=QKV scatter, 1=proj+window_reverse+residual, 2=fc1+GELU, 3=fc2+residual
template<int EPI>
__global__ __launch_bounds__(256, 2)
void gemm_bf16_k(const __bf16* __restrict__ A, const __bf16* __restrict__ Bw,
                 const float* __restrict__ bias, int K, int Nc,
                 const float* __restrict__ fin, float* __restrict__ fout,
                 __bf16* __restrict__ bo0, __bf16* __restrict__ bo1,
                 __bf16* __restrict__ bo2)
{
  __shared__ __align__(16) __bf16 As[BM * LDA];
  __shared__ __align__(16) __bf16 Bs[BN * LDB];

  const int tid  = threadIdx.x;
  const int lane = tid & 31, wid = tid >> 5;
  const int wm = (wid >> 1) * 32, wn = (wid & 1) * 32;
  const int lm = lane & 15, kh = lane >> 4;
  const int m0 = blockIdx.y * BM, n0 = blockIdx.x * BN;

#if USE_TDM
  const unsigned asOff = lds_addr32(As);
#endif

  v8f acc[2][2] = {};

  for (int k0 = 0; k0 < K; k0 += BK) {
    // --- stage A tile (128 x 32) ---
#if USE_TDM
    if (wid == 0)
      tdm_load_2d_bf16(A + (size_t)m0 * K + k0, asOff, BM, BK, K);
#else
#pragma unroll
    for (int t = 0; t < 2; ++t) {
      int c = tid + t * 256;            // 512 16B chunks
      int row = c >> 2, seg = c & 3;
      uint4 d = *(const uint4*)(A + (size_t)(m0 + row) * K + k0 + seg * 8);
      *(uint4*)(As + row * LDA + seg * 8) = d;
    }
#endif
    // --- stage B tile transposed into LDS: Bs[n][k] ---
    {
      int kr = tid >> 3, seg = tid & 7; // 256 16B chunks (32 x 64)
      uint4 d = *(const uint4*)(Bw + (size_t)(k0 + kr) * Nc + n0 + seg * 8);
      unsigned short e[8]; __builtin_memcpy(e, &d, 16);
#pragma unroll
      for (int j = 0; j < 8; ++j) {
        __bf16 bb; __builtin_memcpy(&bb, &e[j], 2);
        Bs[(seg * 8 + j) * LDB + kr] = bb;
      }
    }
    if (k0 + BK < K)   // prefetch next B K-step (global_prefetch_b8)
      __builtin_prefetch(Bw + (size_t)(k0 + BK + (tid >> 3)) * Nc + n0, 0, 0);
#if USE_TDM
    __builtin_amdgcn_s_wait_tensorcnt(0);
#endif
    __syncthreads();

    // --- fragments per ISA 7.12.2 layouts ---
    Frag16 a[2], b[2];
#pragma unroll
    for (int mt = 0; mt < 2; ++mt) {
      const __bf16* p = As + (wm + mt * 16 + lm) * LDA;
      a[mt].u[0] = *(const uint4*)(p + kh * 8);        // K = kh*8 .. +8
      a[mt].u[1] = *(const uint4*)(p + 16 + kh * 8);   // K = 16+kh*8 .. +8
    }
#pragma unroll
    for (int nt = 0; nt < 2; ++nt) {
      const __bf16* p = Bs + (wn + nt * 16 + lm) * LDB;
      b[nt].u[0] = *(const uint4*)(p + kh * 8);        // K = kh*8 .. +8
      b[nt].u[1] = *(const uint4*)(p + 16 + kh * 8);   // K = 16+kh*8 .. +8
    }
#pragma unroll
    for (int mt = 0; mt < 2; ++mt)
#pragma unroll
      for (int nt = 0; nt < 2; ++nt)
        acc[mt][nt] = __builtin_amdgcn_wmma_f32_16x16x32_bf16(
            false, a[mt].v, false, b[nt].v, (short)0, acc[mt][nt], false, false);
    __syncthreads();
  }

  // --- epilogue ---
#pragma unroll
  for (int mt = 0; mt < 2; ++mt) {
#pragma unroll
    for (int nt = 0; nt < 2; ++nt) {
      const int col = n0 + wn + nt * 16 + lm;
      const float bv = bias[col];
#pragma unroll
      for (int r = 0; r < 8; ++r) {
        const int row = m0 + wm + mt * 16 + ((lane < 16) ? r : r + 8);
        float val = acc[mt][nt][r] + bv;
        if constexpr (EPI == 0) {
          // qkv: (B_,N,3,NH,HD) -> q/k/v each (B_,NH,N,HD); fold 1/sqrt(HD) into q
          int t3 = col / 192, rem = col - t3 * 192;
          int head = rem >> 5, d = rem & 31;
          size_t b_ = (size_t)(row >> 6); int n = row & 63;
          size_t off = b_ * 12288 + (size_t)head * 2048 + (size_t)n * 32 + d;
          if (t3 == 0)      bo0[off] = f2bf(val * 0.17677669529663687f);
          else if (t3 == 1) bo1[off] = f2bf(val);
          else              bo2[off] = f2bf(val);
        } else if constexpr (EPI == 1) {
          // proj: window_reverse + roll(+SS) + residual; write fp32 and bf16
          int b = row >> 12, wiw = (row >> 6) & 63, n = row & 63;
          int hs = (wiw >> 3) * 8 + (n >> 3), wsr = (wiw & 7) * 8 + (n & 7);
          int hh = (hs + 4) & 63, wwp = (wsr + 4) & 63;
          size_t oidx = ((size_t)b * 4096 + hh * 64 + wwp) * 192 + col;
          float res = val + fin[oidx];
          fout[oidx] = res;
          bo0[oidx] = f2bf(res);
        } else if constexpr (EPI == 2) {
          // fc1: exact GELU -> bf16 hidden
          float g = 0.5f * val * (1.0f + erff(val * 0.70710678118654752f));
          bo0[(size_t)row * Nc + col] = f2bf(g);
        } else {
          // fc2: + residual -> final fp32 output
          size_t oidx = (size_t)row * Nc + col;
          fout[oidx] = fin[oidx] + val;
        }
      }
    }
  }
}

// ---------------------------------------------------------------------------
// Attention: one block (128 thr = 4 waves) per (window, head).
// S = (q*scale) k^T via WMMA, fp32 softmax w/ on-the-fly rel-pos bias + shift
// mask, then P v via WMMA. Output scattered as (B_, N, NH*HD) bf16.
// q/k tiles staged with async global->LDS (padded per-lane LDS addresses).
// ---------------------------------------------------------------------------
__global__ __launch_bounds__(128, 2)
void attn_k(const __bf16* __restrict__ q, const __bf16* __restrict__ k,
            const __bf16* __restrict__ v, const float* __restrict__ rpb,
            __bf16* __restrict__ xo)
{
  __shared__ __align__(16) __bf16 qs[64 * 40];
  __shared__ __align__(16) __bf16 ks[64 * 40];
  __shared__ __align__(16) __bf16 vt[32 * 72];   // v transposed: [d][m]
  __shared__ __align__(16) __bf16 Pm[64 * 72];   // softmax probs bf16
  __shared__ float Sm[64 * 65];
  __shared__ float rb[225];

  const int tid = threadIdx.x;
  const int bh = blockIdx.x;
  const int b_ = bh / 6, h = bh % 6;
  const int wi = b_ & 63;                         // window index in 8x8 grid
  const size_t base = (size_t)bh * 2048;          // 64 tokens * 32 dims

#if USE_ASYNC
  for (int t = tid; t < 256; t += 128) {          // 256 16B chunks per matrix
    int row = t >> 2, seg = t & 3;
    __builtin_amdgcn_global_load_async_to_lds_b128(
        (v4i*)(q + base + row * 32 + seg * 8),
        (v4i*)(qs + row * 40 + seg * 8), 0, 0);
    __builtin_amdgcn_global_load_async_to_lds_b128(
        (v4i*)(k + base + row * 32 + seg * 8),
        (v4i*)(ks + row * 40 + seg * 8), 0, 0);
  }
#else
  for (int t = tid; t < 512; t += 128) {          // q,k tiles (64x32) as uint2
    int row = t >> 3, seg = t & 7;
    *(uint2*)(qs + row * 40 + seg * 4) = *(const uint2*)(q + base + row * 32 + seg * 4);
    *(uint2*)(ks + row * 40 + seg * 4) = *(const uint2*)(k + base + row * 32 + seg * 4);
  }
#endif

  for (int t = tid; t < 225; t += 128) rb[t] = rpb[t * 6 + h];

  for (int t = tid; t < 2048; t += 128) {         // v transposed scatter
    int row = t >> 5, d = t & 31;
    vt[d * 72 + row] = v[base + t];
  }
#if USE_ASYNC
  __builtin_amdgcn_s_wait_asynccnt(0);
#endif
  __syncthreads();

  const int lane = tid & 31, wid = tid >> 5;
  const int lm = lane & 15, kh = lane >> 4;
  const int arow = wid * 16 + lm;

  // ---- S = q k^T : wave computes rows [16w,16w+16) x all 64 cols ----
  {
    Frag16 a;
    const __bf16* pa = qs + arow * 40;
    a.u[0] = *(const uint4*)(pa + kh * 8);
    a.u[1] = *(const uint4*)(pa + 16 + kh * 8);
#pragma unroll
    for (int t = 0; t < 4; ++t) {
      Frag16 b;
      const __bf16* pb = ks + (t * 16 + lm) * 40;  // B[K=d][n=m] = k[m][d]
      b.u[0] = *(const uint4*)(pb + kh * 8);
      b.u[1] = *(const uint4*)(pb + 16 + kh * 8);
      v8f s = {};
      s = __builtin_amdgcn_wmma_f32_16x16x32_bf16(false, a.v, false, b.v,
                                                  (short)0, s, false, false);
#pragma unroll
      for (int r = 0; r < 8; ++r) {
        int rrow = wid * 16 + ((lane < 16) ? r : r + 8);
        Sm[rrow * 65 + t * 16 + lm] = s[r];
      }
    }
  }
  __syncthreads();

  // ---- bias + shift-mask + softmax (fp32), one row per thread ----
  if (tid < 64) {
    const int row = tid;
    const int ri = row >> 3, rj = row & 7;
    const int gh = (wi >> 3) * 8 + ri, gw = (wi & 7) * 8 + rj;
    const int cr = (gh < 56 ? 0 : (gh < 60 ? 1 : 2)) * 3 +
                   (gw < 56 ? 0 : (gw < 60 ? 1 : 2));
    float mx = -1e30f;
    for (int m = 0; m < 64; ++m) {
      int mi = m >> 3, mj = m & 7;
      int gh2 = (wi >> 3) * 8 + mi, gw2 = (wi & 7) * 8 + mj;
      int cm = (gh2 < 56 ? 0 : (gh2 < 60 ? 1 : 2)) * 3 +
               (gw2 < 56 ? 0 : (gw2 < 60 ? 1 : 2));
      int idx = (ri - mi + 7) * 15 + (rj - mj + 7);
      float val = Sm[row * 65 + m] + rb[idx] + ((cm != cr) ? -100.0f : 0.0f);
      Sm[row * 65 + m] = val;
      mx = fmaxf(mx, val);
    }
    float sum = 0.f;
    for (int m = 0; m < 64; ++m) {
      float e = __expf(Sm[row * 65 + m] - mx);
      Sm[row * 65 + m] = e;
      sum += e;
    }
    float inv = 1.0f / sum;
    for (int m = 0; m < 64; ++m) Pm[row * 72 + m] = f2bf(Sm[row * 65 + m] * inv);
  }
  __syncthreads();

  // ---- out = P v : wave rows [16w,16w+16) x 32 dims, K=64 in 2 steps ----
  v8f o[2] = {};
#pragma unroll
  for (int ks_ = 0; ks_ < 2; ++ks_) {
    Frag16 a;
    const __bf16* pa = Pm + arow * 72 + ks_ * 32;
    a.u[0] = *(const uint4*)(pa + kh * 8);
    a.u[1] = *(const uint4*)(pa + 16 + kh * 8);
#pragma unroll
    for (int dt = 0; dt < 2; ++dt) {
      Frag16 b;
      const __bf16* pb = vt + (dt * 16 + lm) * 72 + ks_ * 32; // B[K=m][n=d]
      b.u[0] = *(const uint4*)(pb + kh * 8);
      b.u[1] = *(const uint4*)(pb + 16 + kh * 8);
      o[dt] = __builtin_amdgcn_wmma_f32_16x16x32_bf16(false, a.v, false, b.v,
                                                      (short)0, o[dt], false, false);
    }
  }
#pragma unroll
  for (int dt = 0; dt < 2; ++dt)
#pragma unroll
    for (int r = 0; r < 8; ++r) {
      int n = wid * 16 + ((lane < 16) ? r : r + 8);
      int d = dt * 16 + lm;
      xo[((size_t)b_ * 64 + n) * 192 + (size_t)h * 32 + d] = f2bf(o[dt][r]);
    }
}

// ---------------------------------------------------------------------------
// Staging kernels
// ---------------------------------------------------------------------------
__global__ void cast_bf16_k(const float* __restrict__ s, __bf16* __restrict__ d, int n) {
  int i = blockIdx.x * 256 + threadIdx.x;
  if (i < n) d[i] = f2bf(s[i]);
}

// x (B,HW,C) fp32 -> roll(-4,-4) + window partition -> xw (B*64, 64, 192) bf16
__global__ void stage_xw_k(const float* __restrict__ x, __bf16* __restrict__ xw) {
  int o = blockIdx.x * 256 + threadIdx.x;        // 25165824 total
  int c = o % 192;
  int rm = o / 192;
  int b = rm >> 12, wiw = (rm >> 6) & 63, n = rm & 63;
  int hs = (wiw >> 3) * 8 + (n >> 3), wsr = (wiw & 7) * 8 + (n & 7);
  int hh = (hs + 4) & 63, wwp = (wsr + 4) & 63;
  xw[o] = f2bf(x[((size_t)b * 4096 + hh * 64 + wwp) * 192 + c]);
}

// ---------------------------------------------------------------------------
extern "C" void kernel_launch(void* const* d_in, const int* in_sizes, int n_in,
                              void* d_out, int out_size, void* d_ws, size_t ws_size,
                              hipStream_t stream)
{
  (void)in_sizes; (void)n_in; (void)out_size; (void)ws_size;
  const float* x     = (const float*)d_in[0];
  const float* rpb   = (const float*)d_in[1];
  const float* qkvw  = (const float*)d_in[2];
  const float* qkvb  = (const float*)d_in[3];
  const float* projw = (const float*)d_in[4];
  const float* projb = (const float*)d_in[5];
  const float* fc1w  = (const float*)d_in[6];
  const float* fc1b  = (const float*)d_in[7];
  const float* fc2w  = (const float*)d_in[8];
  const float* fc2b  = (const float*)d_in[9];

  char* ws = (char*)d_ws;
  size_t off = 0;
  auto take = [&](size_t bytes) {
    char* p = ws + off;
    off += (bytes + 255) & ~(size_t)255;
    return p;
  };

  constexpr size_t MTOT = 131072;                 // B_*N == B*H*W
  __bf16* wqkv  = (__bf16*)take(110592ull * 2);   // 192x576
  __bf16* wproj = (__bf16*)take(36864ull * 2);    // 192x192
  __bf16* wfc1  = (__bf16*)take(147456ull * 2);   // 192x768
  __bf16* wfc2  = (__bf16*)take(147456ull * 2);   // 768x192
  __bf16* xw    = (__bf16*)take(MTOT * 192 * 2);
  __bf16* qb    = (__bf16*)take(MTOT * 192 * 2);
  __bf16* kb    = (__bf16*)take(MTOT * 192 * 2);
  __bf16* vb    = (__bf16*)take(MTOT * 192 * 2);
  __bf16* xat   = (__bf16*)take(MTOT * 192 * 2);
  float*  x2f   = (float*) take(MTOT * 192 * 4);
  __bf16* x2b   = (__bf16*)take(MTOT * 192 * 2);
  __bf16* hml   = xw;  // MLP hidden (131072x768 bf16) overlays dead xw+q+k+v

  cast_bf16_k<<<(110592 + 255) / 256, 256, 0, stream>>>(qkvw, wqkv, 110592);
  cast_bf16_k<<<(36864  + 255) / 256, 256, 0, stream>>>(projw, wproj, 36864);
  cast_bf16_k<<<(147456 + 255) / 256, 256, 0, stream>>>(fc1w, wfc1, 147456);
  cast_bf16_k<<<(147456 + 255) / 256, 256, 0, stream>>>(fc2w, wfc2, 147456);

  stage_xw_k<<<98304, 256, 0, stream>>>(x, xw);

  // QKV: (131072x192) @ (192x576)
  gemm_bf16_k<0><<<dim3(9, 1024), 256, 0, stream>>>(
      xw, wqkv, qkvb, 192, 576, nullptr, nullptr, qb, kb, vb);

  // attention: 2048 windows * 6 heads
  attn_k<<<12288, 128, 0, stream>>>(qb, kb, vb, rpb, xat);

  // proj + window_reverse + unshift + residual
  gemm_bf16_k<1><<<dim3(3, 1024), 256, 0, stream>>>(
      xat, wproj, projb, 192, 192, x, x2f, x2b, nullptr, nullptr);

  // fc1 + GELU
  gemm_bf16_k<2><<<dim3(12, 1024), 256, 0, stream>>>(
      x2b, wfc1, fc1b, 192, 768, nullptr, nullptr, hml, nullptr, nullptr);

  // fc2 + residual -> output
  gemm_bf16_k<3><<<dim3(3, 1024), 256, 0, stream>>>(
      hml, wfc2, fc2b, 768, 192, x2f, (float*)d_out, nullptr, nullptr, nullptr);
}
